// AttentionalGCN_2388001817050
// MI455X (gfx1250) — compile-verified, matrix operand order checked by reading, not run on hardware
//
#include <hip/hip_runtime.h>
#include <hip/hip_bf16.h>

typedef __attribute__((ext_vector_type(16))) _Float16 v16h;
typedef __attribute__((ext_vector_type(8)))  _Float16 v8h;
typedef __attribute__((ext_vector_type(8)))  float    v8f;

#define IN_C  128
#define HID   32
#define HEADS 4
#define OUT_C 64
#define NEG_SLOPE 0.2f

// ---------------- helpers ----------------
__device__ __forceinline__ unsigned flipf(float f) {
    unsigned u = __float_as_uint(f);
    return (u & 0x80000000u) ? ~u : (u | 0x80000000u);
}
__device__ __forceinline__ float unflipf(unsigned k) {
    unsigned u = (k & 0x80000000u) ? (k & 0x7fffffffu) : ~k;
    return __uint_as_float(u);
}
__device__ __forceinline__ float lrelu(float x) {
    return x >= 0.f ? x : NEG_SLOPE * x;
}

// ---------------- utility kernels ----------------
__global__ void gat_fill_b32(unsigned* __restrict__ p, unsigned v, long long n) {
    long long i = (long long)blockIdx.x * blockDim.x + threadIdx.x;
    if (i < n) p[i] = v;
}

__global__ void gat_cvt_f32_f16(const float* __restrict__ in, _Float16* __restrict__ out, long long n) {
    long long i = (long long)blockIdx.x * blockDim.x + threadIdx.x;
    if (i < n) out[i] = (_Float16)in[i];
}

// ---------------- pack B[K,NC] (f32, row-major) into WMMA fragment layout --
// Packed layout: tile (kt, nt) -> 32 lanes x v16h contiguous.
// Lane (r = lane&15, hi = lane>>4) holds column n = nt*16 + r with
// elements 0..7  : K = kt*32 + hi*8 + i
// elements 8..15 : K = kt*32 + 16 + hi*8 + (i-8)
__global__ void gat_pack_b(const float* __restrict__ B, _Float16* __restrict__ Bp,
                           int K, int NC) {
    long long t = (long long)blockIdx.x * blockDim.x + threadIdx.x;
    if (t >= (long long)K * NC) return;
    int i    = (int)(t & 15);
    int lane = (int)((t >> 4) & 31);
    long long tile = t >> 9;            // kt * (NC/16) + nt
    int NT = NC >> 4;
    int kt = (int)(tile / NT), nt = (int)(tile % NT);
    int r = lane & 15, hi = lane >> 4;
    int k = kt * 32 + (i < 8 ? hi * 8 + i : 16 + hi * 8 + (i - 8));
    int n = nt * 16 + r;
    Bp[t] = (_Float16)B[(size_t)k * NC + n];
}

// ---------------- WMMA GEMM: C[M,NC] = A[M,K] * Bpacked, f16 in, f32 out ---
// block = 128 threads (4 waves); grid = (M/16, NC/64); wave w -> n-tile
// (blockIdx.y*4 + w).  K multiple of 32, M multiple of 16, NC multiple of 16.
template <int K, int NC>
__global__ __launch_bounds__(128)
void gat_gemm_wmma(const _Float16* __restrict__ A, const v16h* __restrict__ Bp,
                   float* __restrict__ C, int M) {
    const int wave = threadIdx.x >> 5;
    const int lane = threadIdx.x & 31;
    const int m0 = blockIdx.x * 16;
    const int n0 = (blockIdx.y * 4 + wave) * 16;
    if (n0 >= NC) return;                 // wave-uniform: EXEC stays all-ones
    const int r  = lane & 15;
    const int hi = lane >> 4;
    constexpr int NT = NC / 16;
    const int nt = n0 >> 4;

    const _Float16* arow = A + (size_t)(m0 + r) * K + hi * 8;   // 16B aligned
    v8f acc = {};
#pragma unroll
    for (int kt = 0; kt < K / 32; ++kt) {
        v8h alo = *(const v8h*)(arow + kt * 32);        // K = kt*32 + hi*8 + 0..7
        v8h ahi = *(const v8h*)(arow + kt * 32 + 16);   // K = kt*32 + 16 + hi*8 + 0..7
        v16h a = __builtin_shufflevector(alo, ahi, 0, 1, 2, 3, 4, 5, 6, 7,
                                                   8, 9, 10, 11, 12, 13, 14, 15);
        v16h b = Bp[((size_t)kt * NT + nt) * 32 + lane];
        acc = __builtin_amdgcn_wmma_f32_16x16x32_f16(false, a, false, b, (short)0, acc,
                                                     false, false);
    }
    // C/D layout: VGPR j -> row m0 + hi*8 + j, col n0 + r
#pragma unroll
    for (int j = 0; j < 8; ++j) {
        C[(size_t)(m0 + hi * 8 + j) * NC + (n0 + r)] = acc[j];
    }
}

// ---------------- attention logits: al[n,h] = dot(h[n,h,:], a[h,:]) --------
__global__ void gat_attn_logits(const float* __restrict__ h,
                                const float* __restrict__ a_s, const float* __restrict__ a_d,
                                float* __restrict__ al_s, float* __restrict__ al_d,
                                long long NH, int H, int F) {
    long long t = (long long)blockIdx.x * blockDim.x + threadIdx.x;
    if (t >= NH) return;
    int head = (int)(t % H);
    const float* hp = h + (size_t)t * F;
    const float* as = a_s + (size_t)head * F;
    const float* ad = a_d + (size_t)head * F;
    float ss = 0.f, sd = 0.f;
    for (int i = 0; i < F; ++i) { float v = hp[i]; ss += v * as[i]; sd += v * ad[i]; }
    al_s[t] = ss;
    al_d[t] = sd;
}

// ---------------- edge pass 1: per-dst running max (flipped-uint atomicMax) -
__global__ void gat_edge_max(const int* __restrict__ src, const int* __restrict__ dst,
                             long long E, long long Etot, int H,
                             const float* __restrict__ al_s, const float* __restrict__ al_d,
                             unsigned* __restrict__ mU) {
    long long e = (long long)blockIdx.x * blockDim.x + threadIdx.x;
    if (e >= Etot) return;
    int s, d;
    if (e < E) { s = src[e]; d = dst[e]; if (s == d) return; }   // removed self loop
    else       { s = d = (int)(e - E); }                          // appended self loop
    for (int h = 0; h < H; ++h) {
        float ev = lrelu(al_s[(size_t)s * H + h] + al_d[(size_t)d * H + h]);
        atomicMax(&mU[(size_t)d * H + h], flipf(ev));
    }
}

// ---------------- edge pass 2: softmax denominator ------------------------
__global__ void gat_edge_denom(const int* __restrict__ src, const int* __restrict__ dst,
                               long long E, long long Etot, int H,
                               const float* __restrict__ al_s, const float* __restrict__ al_d,
                               const unsigned* __restrict__ mU, float* __restrict__ denom) {
    long long e = (long long)blockIdx.x * blockDim.x + threadIdx.x;
    if (e >= Etot) return;
    int s, d;
    if (e < E) { s = src[e]; d = dst[e]; if (s == d) return; }
    else       { s = d = (int)(e - E); }
    for (int h = 0; h < H; ++h) {
        float ev = lrelu(al_s[(size_t)s * H + h] + al_d[(size_t)d * H + h]);
        float mv = unflipf(mU[(size_t)d * H + h]);
        atomicAdd(&denom[(size_t)d * H + h], __expf(ev - mv));
    }
}

// ---------------- edge pass 3: weighted scatter (one wave per edge) -------
__global__ void gat_edge_scatter(const int* __restrict__ src, const int* __restrict__ dst,
                                 long long E, long long Etot, int H, int F,
                                 const float* __restrict__ al_s, const float* __restrict__ al_d,
                                 const unsigned* __restrict__ mU, const float* __restrict__ denom,
                                 const float* __restrict__ hfeat, float* __restrict__ out) {
    long long gid = (long long)blockIdx.x * blockDim.x + threadIdx.x;
    long long e = gid >> 5;
    int lane = (int)(gid & 31);
    if (e >= Etot) return;
    int s, d;
    if (e < E) { s = src[e]; d = dst[e]; if (s == d) return; }
    else       { s = d = (int)(e - E); }
    const int HF = H * F;
    const float* hs = hfeat + (size_t)s * HF;
    float*       od = out   + (size_t)d * HF;
    __builtin_prefetch(hs + lane, 0, 0);   // global_prefetch_b8: warm the gathered row
    for (int hf = lane; hf < HF; hf += 32) {
        int head = hf / F;
        float ev = lrelu(al_s[(size_t)s * H + head] + al_d[(size_t)d * H + head]);
        float mv = unflipf(mU[(size_t)d * H + head]);
        float dv = denom[(size_t)d * H + head] + 1e-16f;
        float alpha = __expf(ev - mv) / dv;
        atomicAdd(&od[hf], alpha * hs[hf]);
    }
}

// ---------------- bias + relu + f16 convert (between layers) --------------
__global__ void gat_bias_relu_cvt(const float* __restrict__ acc, const float* __restrict__ bias,
                                  _Float16* __restrict__ out, long long n, int C) {
    long long i = (long long)blockIdx.x * blockDim.x + threadIdx.x;
    if (i >= n) return;
    float v = acc[i] + bias[i % C];
    out[i] = (_Float16)(v > 0.f ? v : 0.f);
}

// ---------------- final bias add ------------------------------------------
__global__ void gat_bias_add(float* __restrict__ out, const float* __restrict__ bias,
                             long long n, int C) {
    long long i = (long long)blockIdx.x * blockDim.x + threadIdx.x;
    if (i >= n) return;
    out[i] += bias[i % C];
}

// ============================ launch ======================================
extern "C" void kernel_launch(void* const* d_in, const int* in_sizes, int n_in,
                              void* d_out, int out_size, void* d_ws, size_t ws_size,
                              hipStream_t stream) {
    const float* x      = (const float*)d_in[0];
    const int*   ei     = (const int*)  d_in[1];
    const float* W1     = (const float*)d_in[2];
    const float* a_src1 = (const float*)d_in[3];
    const float* a_dst1 = (const float*)d_in[4];
    const float* b1     = (const float*)d_in[5];
    const float* W2     = (const float*)d_in[6];
    const float* a_src2 = (const float*)d_in[7];
    const float* a_dst2 = (const float*)d_in[8];
    const float* b2     = (const float*)d_in[9];
    float* out = (float*)d_out;

    const long long N    = in_sizes[0] / IN_C;     // 50000
    const long long E    = in_sizes[1] / 2;        // 800000
    const long long Etot = E + N;                  // with appended self loops
    const int HF1 = HEADS * HID;                   // 128

    const int* src = ei;
    const int* dst = ei + E;

    // workspace layout (256B aligned)
    char* ws = (char*)d_ws;
    size_t off = 0;
    auto take = [&](size_t bytes) { char* p = ws + off; off = (off + bytes + 255) & ~(size_t)255; return p; };
    _Float16* xh    = (_Float16*)take((size_t)N * IN_C * 2);   // x f16, reused as relu(h1) f16
    _Float16* w1p   = (_Float16*)take((size_t)IN_C * HF1 * 2); // packed WMMA fragments
    _Float16* w2p   = (_Float16*)take((size_t)HF1 * OUT_C * 2);
    float*    hbuf  = (float*)   take((size_t)N * HF1 * 4);    // h1 (layer1), h2 (layer2)
    float*    acc1  = (float*)   take((size_t)N * HF1 * 4);    // layer1 aggregation
    float*    al_s  = (float*)   take((size_t)N * HEADS * 4);
    float*    al_d  = (float*)   take((size_t)N * HEADS * 4);
    unsigned* mU    = (unsigned*)take((size_t)N * HEADS * 4);
    float*    denom = (float*)   take((size_t)N * HEADS * 4);
    (void)ws_size;

    const int TB = 256;
    auto blocks = [&](long long n) { return (unsigned)((n + TB - 1) / TB); };

    // --- conversions / packing ---
    gat_cvt_f32_f16<<<blocks((long long)N * IN_C), TB, 0, stream>>>(x, xh, (long long)N * IN_C);
    gat_pack_b<<<blocks(IN_C * HF1),  TB, 0, stream>>>(W1, w1p, IN_C, HF1);
    gat_pack_b<<<blocks(HF1 * OUT_C), TB, 0, stream>>>(W2, w2p, HF1, OUT_C);

    // ================= layer 1 (H=4, F=32) =================
    {
        dim3 g((unsigned)(N / 16), HF1 / 64);
        gat_gemm_wmma<IN_C, HF1><<<g, 128, 0, stream>>>(xh, (const v16h*)w1p, hbuf, (int)N);
    }
    gat_attn_logits<<<blocks(N * HEADS), TB, 0, stream>>>(hbuf, a_src1, a_dst1, al_s, al_d,
                                                          N * HEADS, HEADS, HID);
    gat_fill_b32<<<blocks(N * HEADS), TB, 0, stream>>>(mU, 0u, N * HEADS);
    gat_fill_b32<<<blocks(N * HEADS), TB, 0, stream>>>((unsigned*)denom, 0u, N * HEADS);
    gat_fill_b32<<<blocks(N * HF1),   TB, 0, stream>>>((unsigned*)acc1, 0u, N * HF1);
    gat_edge_max    <<<blocks(Etot), TB, 0, stream>>>(src, dst, E, Etot, HEADS, al_s, al_d, mU);
    gat_edge_denom  <<<blocks(Etot), TB, 0, stream>>>(src, dst, E, Etot, HEADS, al_s, al_d, mU, denom);
    gat_edge_scatter<<<blocks(Etot * 32), TB, 0, stream>>>(src, dst, E, Etot, HEADS, HID,
                                                           al_s, al_d, mU, denom, hbuf, acc1);
    // bias + relu -> f16 (reuse xh)
    gat_bias_relu_cvt<<<blocks(N * HF1), TB, 0, stream>>>(acc1, b1, xh, N * HF1, HF1);

    // ================= layer 2 (H=1, F=64) =================
    {
        dim3 g((unsigned)(N / 16), 1);  // 4 waves cover the 4 n-tiles of OUT_C=64
        gat_gemm_wmma<HF1, OUT_C><<<g, 128, 0, stream>>>(xh, (const v16h*)w2p, hbuf, (int)N);
    }
    gat_attn_logits<<<blocks(N), TB, 0, stream>>>(hbuf, a_src2, a_dst2, al_s, al_d, N, 1, OUT_C);
    gat_fill_b32<<<blocks(N),         TB, 0, stream>>>(mU, 0u, N);
    gat_fill_b32<<<blocks(N),         TB, 0, stream>>>((unsigned*)denom, 0u, N);
    gat_fill_b32<<<blocks(N * OUT_C), TB, 0, stream>>>((unsigned*)out, 0u, N * OUT_C);
    gat_edge_max    <<<blocks(Etot), TB, 0, stream>>>(src, dst, E, Etot, 1, al_s, al_d, mU);
    gat_edge_denom  <<<blocks(Etot), TB, 0, stream>>>(src, dst, E, Etot, 1, al_s, al_d, mU, denom);
    gat_edge_scatter<<<blocks(Etot * 32), TB, 0, stream>>>(src, dst, E, Etot, 1, OUT_C,
                                                           al_s, al_d, mU, denom, hbuf, out);
    gat_bias_add<<<blocks(N * OUT_C), TB, 0, stream>>>(out, b2, N * OUT_C, OUT_C);
}
